// Re_SS_Cos_Loss_59837484368379
// MI455X (gfx1250) — matrix operand order repaired and checked by reading, not compiled, not run on hardware
//
#include <hip/hip_runtime.h>
#include <hip/hip_bf16.h>
#include <math.h>

#define ALPHA 0.5f
#define EPS   1e-07f

#define B_   2
#define C_   128
#define N_   4096            // 16*16*16
#define NT_  256             // N_/16 tiles per dimension

typedef __attribute__((ext_vector_type(8)))  float   v8f;
typedef __attribute__((ext_vector_type(8)))  __bf16  v8bf;
typedef __attribute__((ext_vector_type(16))) __bf16  v16bf;

// ---------------------------------------------------------------------------
// Workspace layout (bytes):
//   S  bf16 [B][N][C]   : 0          .. 2 MiB
//   Q  bf16 [B][N][C]   : 2 MiB      .. 4 MiB
//   sn f32  [B][N]      : 4 MiB      .. +32 KiB
//   qn f32  [B][N]      : ...        .. +32 KiB
//   accum (2 f32 + 2 u32): tail
// ---------------------------------------------------------------------------
#define OFF_S   ((size_t)0)
#define OFF_Q   ((size_t)(B_) * N_ * C_ * 2)
#define OFF_SN  (OFF_Q + (size_t)(B_) * N_ * C_ * 2)
#define OFF_QN  (OFF_SN + (size_t)(B_) * N_ * 4)
#define OFF_ACC (OFF_QN + (size_t)(B_) * N_ * 4)

// ---------------------------------------------------------------------------
// Zero the scalar accumulators (harness poisons ws with 0xAA).
// ---------------------------------------------------------------------------
__global__ void initacc(float* sums, unsigned* cnts) {
    sums[0] = 0.f; sums[1] = 0.f;
    cnts[0] = 0u;  cnts[1] = 0u;
}

// ---------------------------------------------------------------------------
// prep: supp = inputs*t_pos, query = inputs*t_neg (fp32), column norms over C
// in fp32, values stored as bf16 in [b][n][c] layout (C contiguous).
// One thread per (b, n); reads of inputs[b,c,n] are coalesced across threads.
// ---------------------------------------------------------------------------
__global__ void __launch_bounds__(256) prep_kernel(
    const float* __restrict__ inputs, const float* __restrict__ target,
    __bf16* __restrict__ S, __bf16* __restrict__ Q,
    float* __restrict__ sn, float* __restrict__ qn)
{
    int tid = blockIdx.x * blockDim.x + threadIdx.x;     // 0 .. B_*N_-1
    if (tid >= B_ * N_) return;
    int b = tid >> 12;                                   // N_ = 4096 = 2^12
    int n = tid & (N_ - 1);

    float tneg = target[((size_t)b * 2 + 0) * N_ + n];
    float tpos = target[((size_t)b * 2 + 1) * N_ + n];

    const float* in_b = inputs + (size_t)b * C_ * N_;
    __bf16* Srow = S + (size_t)tid * C_;
    __bf16* Qrow = Q + (size_t)tid * C_;

    float ss = 0.f, qq = 0.f;
    #pragma unroll 4
    for (int c = 0; c < C_; ++c) {
        float v = in_b[(size_t)c * N_ + n];
        float s = v * tpos;
        float q = v * tneg;
        ss += s * s;
        qq += q * q;
        Srow[c] = (__bf16)s;
        Qrow[c] = (__bf16)q;
    }
    sn[tid] = sqrtf(ss);
    qn[tid] = sqrtf(qq);
}

// ---------------------------------------------------------------------------
// Tile loaders: A and B share a contiguous k order (not the ISA table order; any shared order is valid).
//
// A-matrix 16x32 (MxK), lane m = lane&15, half = lane>>4:
//   lanes 0-15 hold K = c0+0..15, lanes 16-31 hold K = c0+16..31 (contiguous)
//   => one 32-byte (16 x bf16) load at element offset half*16.
//
// B-matrix 32x16 (KxN), column n = lane&15:
//   lanes 0-15 hold K = c0+0..15, lanes 16-31 hold K = c0+16..31 (contiguous)
//   => one 32-byte (16 x bf16) load.
// Rows are C_=128 bf16 = 256 bytes, so all loads are >=16-byte aligned.
// ---------------------------------------------------------------------------
__device__ __forceinline__ v16bf load_tileA(const __bf16* __restrict__ base,
                                            int row0, int m, int half, int c0) {
    const __bf16* p = base + (size_t)(row0 + m) * C_ + c0 + half * 16;
    return *(const v16bf*)(p);
}

__device__ __forceinline__ v16bf load_tileB(const __bf16* __restrict__ base,
                                            int row0, int n, int half, int c0) {
    const __bf16* p = base + (size_t)(row0 + n) * C_ + c0 + half * 16;
    return *(const v16bf*)(p);
}

// ---------------------------------------------------------------------------
// main: one wave per 16x16 output tile (i0,j0). 4 K-chunks x 3 WMMAs (SS, QQ,
// SQ), then a register epilogue. No division anywhere in the hot loop:
//   * nonzero counts use __ballot + popcount: one v_cmp feeding SALU bcnt/add
//     (co-executes with the VALU/WMMA pipes); for cos_diff the test is on the
//     raw accumulator since den > 0 => cos!=0 <=> num!=0.
//   * cos = num * rcp(n_i+eps) * rcp(n_j+eps): one v_rcp_f32 per norm, with
//     row reciprocals hoisted per wave and column reciprocals once per tile.
//     (eps placement changes the denominator by ~4e-9 relative, far below the
//     bf16 quantization; it keeps zero-norm rows finite like the reference.)
// C/D layout: lanes 0-15 -> col N = lane, VGPR r -> row M = r;
//             lanes 16-31 -> col N = lane-16, VGPR r -> row M = r + 8.
// ---------------------------------------------------------------------------
__global__ void __launch_bounds__(256) gram_loss_main(
    const __bf16* __restrict__ S, const __bf16* __restrict__ Q,
    const float* __restrict__ sn, const float* __restrict__ qn,
    float* __restrict__ sums, unsigned* __restrict__ cnts)
{
    const int lane = threadIdx.x & 31;
    const int wave = threadIdx.x >> 5;      // 8 waves per block
    const int m    = lane & 15;
    const int half = lane >> 4;

    const int b  = blockIdx.x >> 8;         // NT_ = 256 i-tiles per batch
    const int i0 = (blockIdx.x & (NT_ - 1)) * 16;

    const __bf16* Sb = S + (size_t)b * N_ * C_;
    const __bf16* Qb = Q + (size_t)b * N_ * C_;
    const float*  snb = sn + (size_t)b * N_;
    const float*  qnb = qn + (size_t)b * N_;

    // Hoisted per-lane reciprocal row norms for the i dimension (M = 8*half+r)
    float inv_sni[8], inv_qni[8];
    #pragma unroll
    for (int r = 0; r < 8; ++r) {
        inv_sni[r] = __builtin_amdgcn_rcpf(snb[i0 + 8 * half + r] + EPS);
        inv_qni[r] = __builtin_amdgcn_rcpf(qnb[i0 + 8 * half + r] + EPS);
    }

    float    sum_diff = 0.f, sum_same = 0.f;
    unsigned cnt_diff = 0u,  cnt_same = 0u;   // wave-uniform (ballot counts)

    for (int jt = wave; jt < NT_; jt += 8) {      // uniform: EXEC stays all-1s
        const int j0 = jt * 16;
        v8f acc_ss = {}; v8f acc_qq = {}; v8f acc_sq = {};

        #pragma unroll
        for (int kk = 0; kk < 4; ++kk) {
            const int c0 = kk * 32;
            v16bf aS = load_tileA(Sb, i0, m, half, c0);
            v16bf aQ = load_tileA(Qb, i0, m, half, c0);
            v16bf bS = load_tileB(Sb, j0, m, half, c0);
            v16bf bQ = load_tileB(Qb, j0, m, half, c0);
            acc_ss = __builtin_amdgcn_wmma_f32_16x16x32_bf16(
                         false, aS, false, bS, (short)0, acc_ss, false, false);
            acc_qq = __builtin_amdgcn_wmma_f32_16x16x32_bf16(
                         false, aQ, false, bQ, (short)0, acc_qq, false, false);
            acc_sq = __builtin_amdgcn_wmma_f32_16x16x32_bf16(
                         false, aS, false, bQ, (short)0, acc_sq, false, false);
        }

        const float inv_snj = __builtin_amdgcn_rcpf(snb[j0 + m] + EPS);
        const float inv_qnj = __builtin_amdgcn_rcpf(qnb[j0 + m] + EPS);

        float tdiff = 0.f;
        #pragma unroll
        for (int r = 0; r < 8; ++r) {
            // ---- cos_diff: sum factored (x inv_qnj once), count on raw num
            tdiff = fmaf(acc_sq[r], inv_sni[r], tdiff);
            cnt_diff += (unsigned)__popcll(__ballot(acc_sq[r] != 0.0f));
            // ---- cos_same: need the per-element value for the nnz test
            float u = acc_ss[r] * inv_sni[r];
            float v = acc_qq[r] * inv_qni[r];
            float s_elem = fmaf(u, inv_snj, v * inv_qnj);
            sum_same += s_elem;
            cnt_same += (unsigned)__popcll(__ballot(s_elem != 0.0f));
        }
        sum_diff = fmaf(tdiff, inv_qnj, sum_diff);
    }

    // wave32 butterfly reduction (float sums only; counts are wave-uniform)
    #pragma unroll
    for (int off = 16; off > 0; off >>= 1) {
        sum_diff += __shfl_xor(sum_diff, off);
        sum_same += __shfl_xor(sum_same, off);
    }
    if (lane == 0) {
        atomicAdd(&sums[0], sum_diff);
        atomicAdd(&sums[1], sum_same);
        atomicAdd(&cnts[0], cnt_diff);
        atomicAdd(&cnts[1], cnt_same);
    }
}

// ---------------------------------------------------------------------------
// finalize: scalar combine.
// ---------------------------------------------------------------------------
__global__ void finalize_kernel(const float* __restrict__ sums,
                                const unsigned* __restrict__ cnts,
                                float* __restrict__ out)
{
    float mean_diff = sums[0] / ((float)cnts[0] + EPS);
    float mean_same = sums[1] / ((float)cnts[1] + EPS);
    out[0] = ALPHA * mean_diff + (1.0f - ALPHA) * (1.0f - mean_same);
}

extern "C" void kernel_launch(void* const* d_in, const int* in_sizes, int n_in,
                              void* d_out, int out_size, void* d_ws, size_t ws_size,
                              hipStream_t stream) {
    const float* inputs = (const float*)d_in[0];   // (2,128,16,16,16) f32
    const float* target = (const float*)d_in[1];   // (2,2,16,16,16)   f32
    float* out = (float*)d_out;                    // scalar f32

    char* ws = (char*)d_ws;
    __bf16* S    = (__bf16*)(ws + OFF_S);
    __bf16* Q    = (__bf16*)(ws + OFF_Q);
    float*  snv  = (float*)(ws + OFF_SN);
    float*  qnv  = (float*)(ws + OFF_QN);
    float*    sums = (float*)(ws + OFF_ACC);
    unsigned* cnts = (unsigned*)(ws + OFF_ACC + 2 * sizeof(float));

    initacc<<<1, 1, 0, stream>>>(sums, cnts);

    prep_kernel<<<(B_ * N_ + 255) / 256, 256, 0, stream>>>(
        inputs, target, S, Q, snv, qnv);

    gram_loss_main<<<B_ * NT_, 256, 0, stream>>>(S, Q, snv, qnv, sums, cnts);

    finalize_kernel<<<1, 1, 0, stream>>>(sums, cnts, out);
}
